// SARoutingBlock_56547539419737
// MI455X (gfx1250) — matrix operand order repaired and checked by reading, not compiled, not run on hardware
//
#include <hip/hip_runtime.h>

#define B_  8
#define N_  512
#define D_  1024
#define H_  8
#define DH_ 128
#define L_  4
#define M_  (B_ * N_)   // 4096 rows for the big GEMMs

typedef __bf16 bf16_t;
typedef __attribute__((ext_vector_type(16))) __bf16 v16bf;
typedef __attribute__((ext_vector_type(8)))  __bf16 v8bf;
typedef __attribute__((ext_vector_type(8)))  float  v8f;

// ---------------------------------------------------------------- TDM glue
#if defined(__has_builtin)
#if __has_builtin(__builtin_amdgcn_tensor_load_to_lds) && __has_builtin(__builtin_amdgcn_s_wait_tensorcnt)
#define HAVE_TDM 1
#endif
#endif
#ifndef HAVE_TDM
#define HAVE_TDM 0
#endif

#if HAVE_TDM
typedef __attribute__((ext_vector_type(4))) unsigned int v4u;
typedef __attribute__((ext_vector_type(8))) int v8i_t;
typedef __attribute__((ext_vector_type(4))) int v4i_t;
typedef __attribute__((address_space(3))) const char* lds_cptr_t;

static __device__ inline unsigned lds_off(const void* p) {
    return (unsigned)(unsigned long long)(lds_cptr_t)p;
}

// 2D tile load: tile1 rows of tile0 bf16 elements, row stride stride0 (elems),
// packed contiguously into LDS as [tile1][tile0]. D# per ISA ch.8.
static __device__ inline void tdm_load_2d(bf16_t* ldst, const bf16_t* gsrc,
                                          unsigned tile0, unsigned tile1,
                                          unsigned long long stride0) {
    const unsigned long long ga = (unsigned long long)gsrc;
    v4u g0;
    g0[0] = 1u;                                              // count=1, user D#
    g0[1] = lds_off(ldst);                                   // lds_addr (bytes)
    g0[2] = (unsigned)ga;                                    // global_addr[31:0]
    g0[3] = (unsigned)((ga >> 32) & 0x01FFFFFFull) | 0x80000000u; // [56:32] | type=2
    v8i_t g1;
    g1[0] = 0x00010000;                                      // wg_mask=0, data_size=2B
    g1[1] = (int)((tile0 & 0xFFFFu) << 16);                  // tensor_dim0 = tile0
    g1[2] = (int)(((tile0 >> 16) & 0xFFFFu) | ((tile1 & 0xFFFFu) << 16)); // tensor_dim1
    g1[3] = (int)(((tile1 >> 16) & 0xFFFFu) | ((tile0 & 0xFFFFu) << 16)); // tile_dim0
    g1[4] = (int)(tile1 & 0xFFFFu);                          // tile_dim1, tile_dim2=0
    g1[5] = (int)(unsigned)(stride0 & 0xFFFFFFFFull);        // dim0_stride[31:0]
    g1[6] = (int)(unsigned)((stride0 >> 32) & 0xFFFFull);    // dim0_stride[47:32]
    g1[7] = 0;
    v4i_t z4;
    z4[0] = 0; z4[1] = 0; z4[2] = 0; z4[3] = 0;
    v8i_t z8;
#pragma unroll
    for (int i = 0; i < 8; ++i) z8[i] = 0;
    // 6-arg form (amdgpu-toolchain / clang-23): (g0, g1, g2, g3, g4, cpol)
    __builtin_amdgcn_tensor_load_to_lds(g0, g1, z4, z4, z8, 0);
}
#endif

// Stage a [rows][rowElems] bf16 tile (row stride ldsrc elems) into LDS.
static __device__ inline void stage_tile(bf16_t* dst, const bf16_t* src,
                                         int rows, int rowElems, long ldsrc,
                                         int nthreads) {
#if HAVE_TDM
    (void)nthreads;
    if (threadIdx.x == 0)
        tdm_load_2d(dst, src, (unsigned)rowElems, (unsigned)rows,
                    (unsigned long long)ldsrc);
#else
    const int chunks = rows * (rowElems >> 3);
    for (int idx = threadIdx.x; idx < chunks; idx += nthreads) {
        const int r = idx / (rowElems >> 3);
        const int c = idx - r * (rowElems >> 3);
        *(v8bf*)(dst + r * rowElems + c * 8) =
            *(const v8bf*)(src + (long)r * ldsrc + c * 8);
    }
#endif
}

static __device__ inline void tile_wait() {
#if HAVE_TDM
    __builtin_amdgcn_s_wait_tensorcnt((short)0);
#endif
}

// ---------------------------------------------------------------- WMMA glue
static __device__ inline v8f zero8() {
    v8f z;
#pragma unroll
    for (int i = 0; i < 8; ++i) z[i] = 0.0f;
    return z;
}

static __device__ inline v8f wmma_bf16f32(v16bf a, v16bf b, v8f c) {
    return __builtin_amdgcn_wmma_f32_16x16x32_bf16(
        false, a, false, b, (short)0, c, false, false);
}

// A fragment (16x32): lanes 0-15 row M=lane hold K 0..7 & 16..23,
// lanes 16-31 row M=lane-16 hold K 8..15 & 24..31 (ISA 7.12.2).
static __device__ inline v16bf frag_a(const bf16_t* __restrict__ tile, int ld) {
    const int lane = threadIdx.x & 31;
    const int r    = lane & 15;
    const int kh   = (lane >> 4) << 3;
    const bf16_t* rp = tile + (size_t)r * ld + kh;
    v8bf lo = *(const v8bf*)(rp);
    v8bf hi = *(const v8bf*)(rp + 16);
    v16bf o;
#pragma unroll
    for (int i = 0; i < 8; ++i) { o[i] = lo[i]; o[i + 8] = hi[i]; }
    return o;
}

// B fragment (32x16): lane%16 selects column N (= source row), lanes 0-15
// hold K 0..7 & 16..23, lanes 16-31 hold K 8..15 & 24..31. Source row-major [Ncols, K].
static __device__ inline v16bf frag_b(const bf16_t* __restrict__ tile, int ld) {
    const int lane = threadIdx.x & 31;
    const int n    = lane & 15;
    const int kh   = (lane >> 4) << 3;
    const bf16_t* rp = tile + (size_t)n * ld + kh;
    v8bf lo = *(const v8bf*)(rp);
    v8bf hi = *(const v8bf*)(rp + 16);
    v16bf o;
#pragma unroll
    for (int i = 0; i < 8; ++i) { o[i] = lo[i]; o[i + 8] = hi[i]; }
    return o;
}

// ---------------------------------------------------------------- convert
__global__ __launch_bounds__(256) void f32_to_bf16_kernel(
    const float* __restrict__ src, bf16_t* __restrict__ dst, int n) {
    int i = blockIdx.x * blockDim.x + threadIdx.x;
    int stride = gridDim.x * blockDim.x;
    for (; i < n; i += stride) dst[i] = (bf16_t)src[i];
}

// ------------------------------------------------- shared GEMM main loop
// Block: 256 thr / 8 waves; tile 128(M) x 64(N); K swept in 32-steps.
// B tiles TDM-staged to LDS (double-buffered), A frags double-buffered in regs.
static __device__ inline void gemm_128x64_body(
    const bf16_t* __restrict__ Arow,   // X + m0*D_
    const bf16_t* __restrict__ Wt,     // W + n0*D_
    bf16_t* Bt0, bf16_t* Bt1, v8f acc[4])
{
    stage_tile(Bt0, Wt, 64, 32, D_, 256);
    tile_wait();
    __syncthreads();

    v16bf a_cur = frag_a(Arow, D_);
    for (int i = 0; i < D_ / 32; ++i) {
        const int kk = i * 32;
        bf16_t* curB = (i & 1) ? Bt1 : Bt0;
        bf16_t* nxtB = (i & 1) ? Bt0 : Bt1;
        const bool more = (i + 1) < (D_ / 32);
        if (more) stage_tile(nxtB, Wt + kk + 32, 64, 32, D_, 256);
        v16bf a_nxt = a_cur;
        if (more) a_nxt = frag_a(Arow + kk + 32, D_);
#pragma unroll
        for (int t = 0; t < 4; ++t) {
            v16bf b = frag_b(curB + t * 16 * 32, 32);
            acc[t] = wmma_bf16f32(a_cur, b, acc[t]);
        }
        a_cur = a_nxt;
        tile_wait();
        __syncthreads();
    }
}

// ------------------------------------------------------- QKV projections
__global__ __launch_bounds__(256) void proj_kernel(
    const bf16_t* __restrict__ qb, const bf16_t* __restrict__ kb,
    const bf16_t* __restrict__ vb,
    const bf16_t* __restrict__ wq, const bf16_t* __restrict__ wk,
    const bf16_t* __restrict__ wv,
    const float* __restrict__ bq, const float* __restrict__ bk,
    const float* __restrict__ bv,
    bf16_t* __restrict__ Qp, bf16_t* __restrict__ Kp, bf16_t* __restrict__ VpT)
{
    __shared__ __align__(16) bf16_t Bt[2][64 * 32];   // 8 KB

    const int which = blockIdx.z;
    const bf16_t* X    = which == 0 ? qb : which == 1 ? kb : vb;
    const bf16_t* W    = which == 0 ? wq : which == 1 ? wk : wv;
    const float*  bias = which == 0 ? bq : which == 1 ? bk : bv;
    const float scale  = which == 0 ? 0.08838834764831845f : 1.0f; // 1/sqrt(128)

    const int wave = threadIdx.x >> 5;
    const int lane = threadIdx.x & 31;
    const int m0 = blockIdx.y * 128 + wave * 16;
    const int n0 = blockIdx.x * 64;

    v8f acc[4];
#pragma unroll
    for (int t = 0; t < 4; ++t) acc[t] = zero8();

    gemm_128x64_body(X + (size_t)m0 * D_, W + (size_t)n0 * D_,
                     Bt[0], Bt[1], acc);

    const int colb = n0 + (lane & 15);
    const int rsel = (lane >> 4) << 3;
#pragma unroll
    for (int t = 0; t < 4; ++t) {
        const int col = colb + t * 16;
        const float bias_c = bias[col];
#pragma unroll
        for (int i = 0; i < 8; ++i) {
            const int m = m0 + i + rsel;
            const float val = (acc[t][i] + bias_c) * scale;
            if (which == 0) {
                Qp[(size_t)m * D_ + col] = (bf16_t)val;
            } else if (which == 1) {
                Kp[(size_t)m * D_ + col] = (bf16_t)val;
            } else {
                const int bb = m >> 9, nn = m & (N_ - 1);
                VpT[((size_t)bb * D_ + col) * N_ + nn] = (bf16_t)val;
            }
        }
    }
}

// ------------------------------------------------------------- attention
// Block = one (b,h) x 64 queries; 4 waves x 16 queries.
// K and V operand tiles are TDM-staged to LDS, shared by all 4 waves.
__global__ __launch_bounds__(128) void attn_kernel(
    const bf16_t* __restrict__ Qp, const bf16_t* __restrict__ Kp,
    const bf16_t* __restrict__ VpT,
    const unsigned char* __restrict__ masks, const float* __restrict__ alphas,
    bf16_t* __restrict__ atted)
{
    __shared__ __align__(16) bf16_t S[4][16][N_];       // 64 KB
    __shared__ __align__(16) bf16_t Kt[2][16 * DH_];    //  8 KB
    __shared__ __align__(16) bf16_t Vt[2][DH_ * 32];    // 16 KB

    const int wave = threadIdx.x >> 5;
    const int lane = threadIdx.x & 31;
    const int bh = blockIdx.y;
    const int b = bh >> 3, h = bh & 7;
    const int q0 = blockIdx.x * 64 + wave * 16;
    const int rsel = (lane >> 4) << 3;

    const bf16_t* qtile = Qp + ((size_t)(b * N_ + q0)) * D_ + h * DH_;
    const bf16_t* kbase = Kp + (size_t)b * N_ * D_ + h * DH_;

    // ---- phase 1: scores S = Q*K^T (Q pre-scaled by 1/sqrt(DH))
    v16bf qfrag[4];
#pragma unroll
    for (int t = 0; t < 4; ++t) qfrag[t] = frag_a(qtile + t * 32, D_);

    stage_tile(Kt[0], kbase, 16, DH_, D_, 128);
    tile_wait();
    __syncthreads();

    for (int kt = 0; kt < N_ / 16; ++kt) {
        const bool more = (kt + 1) < (N_ / 16);
        if (more)
            stage_tile(Kt[(kt + 1) & 1], kbase + (size_t)(kt + 1) * 16 * D_,
                       16, DH_, D_, 128);
        v8f acc = zero8();
        const bf16_t* ktile = Kt[kt & 1];
#pragma unroll
        for (int t = 0; t < 4; ++t)
            acc = wmma_bf16f32(qfrag[t], frag_b(ktile + t * 32, DH_), acc);

        const int col = kt * 16 + (lane & 15);
#pragma unroll
        for (int i = 0; i < 8; ++i)
            S[wave][i + rsel][col] = (bf16_t)acc[i];
        tile_wait();
        __syncthreads();
    }

    // ---- phase 2: 4-mask softmax combined by alpha (in place, per wave)
    float al[L_];
#pragma unroll
    for (int l = 0; l < L_; ++l) al[l] = alphas[b * L_ + l];

    unsigned mk[L_];   // key = lane + 32*c
#pragma unroll
    for (int l = 0; l < L_; ++l) {
        unsigned bits = 0;
        const unsigned char* mp = masks + ((size_t)l * B_ + b) * N_;
#pragma unroll
        for (int c = 0; c < 16; ++c)
            bits |= (mp[lane + 32 * c] ? 1u : 0u) << c;
        mk[l] = bits;
    }

    for (int r = 0; r < 16; ++r) {
        float s[16];
#pragma unroll
        for (int c = 0; c < 16; ++c) s[c] = (float)S[wave][r][lane + 32 * c];

        float mx[L_];
#pragma unroll
        for (int l = 0; l < L_; ++l) {
            float m = -3.0e38f;
#pragma unroll
            for (int c = 0; c < 16; ++c) {
                const float v = ((mk[l] >> c) & 1u) ? -1.0e9f : s[c];
                m = fmaxf(m, v);
            }
#pragma unroll
            for (int off = 16; off >= 1; off >>= 1)
                m = fmaxf(m, __shfl_xor(m, off, 32));
            mx[l] = m;
        }

        float w[L_];
#pragma unroll
        for (int l = 0; l < L_; ++l) {
            float sum = 0.0f;
#pragma unroll
            for (int c = 0; c < 16; ++c) {
                const float v = ((mk[l] >> c) & 1u) ? -1.0e9f : s[c];
                sum += __expf(v - mx[l]);
            }
#pragma unroll
            for (int off = 16; off >= 1; off >>= 1)
                sum += __shfl_xor(sum, off, 32);
            w[l] = al[l] / sum;
        }

#pragma unroll
        for (int c = 0; c < 16; ++c) {
            float att = 0.0f;
#pragma unroll
            for (int l = 0; l < L_; ++l) {
                const float v = ((mk[l] >> c) & 1u) ? -1.0e9f : s[c];
                att += w[l] * __expf(v - mx[l]);
            }
            S[wave][r][lane + 32 * c] = (bf16_t)att;
        }
    }

    // ---- phase 3: atted = att * V  (K = 512 keys, Nout = DH)
    v8f oacc[8];
#pragma unroll
    for (int t = 0; t < 8; ++t) oacc[t] = zero8();

    const bf16_t* vbase = VpT + ((size_t)b * D_ + h * DH_) * N_;
    stage_tile(Vt[0], vbase, DH_, 32, N_, 128);
    tile_wait();
    __syncthreads();

    for (int i = 0; i < N_ / 32; ++i) {
        const int kk = i * 32;
        const bool more = (i + 1) < (N_ / 32);
        if (more)
            stage_tile(Vt[(i + 1) & 1], vbase + kk + 32, DH_, 32, N_, 128);
        v16bf a = frag_a(&S[wave][0][0] + kk, N_);
        const bf16_t* vt = Vt[i & 1];
#pragma unroll
        for (int t = 0; t < 8; ++t)
            oacc[t] = wmma_bf16f32(a, frag_b(vt + t * 16 * 32, 32), oacc[t]);
        tile_wait();
        __syncthreads();
    }

    bf16_t* obase = atted + ((size_t)(b * N_ + q0)) * D_ + h * DH_;
#pragma unroll
    for (int t = 0; t < 8; ++t) {
        const int col = t * 16 + (lane & 15);
#pragma unroll
        for (int i = 0; i < 8; ++i)
            obase[(size_t)(i + rsel) * D_ + col] = (bf16_t)oacc[t][i];
    }
}

// ----------------------------------------------------- output projection
__global__ __launch_bounds__(256) void outproj_kernel(
    const bf16_t* __restrict__ A, const bf16_t* __restrict__ Wm,
    const float* __restrict__ bm, float* __restrict__ out)
{
    __shared__ __align__(16) bf16_t Bt[2][64 * 32];

    const int wave = threadIdx.x >> 5;
    const int lane = threadIdx.x & 31;
    const int m0 = blockIdx.y * 128 + wave * 16;
    const int n0 = blockIdx.x * 64;

    v8f acc[4];
#pragma unroll
    for (int t = 0; t < 4; ++t) acc[t] = zero8();

    gemm_128x64_body(A + (size_t)m0 * D_, Wm + (size_t)n0 * D_,
                     Bt[0], Bt[1], acc);

    const int colb = n0 + (lane & 15);
    const int rsel = (lane >> 4) << 3;
#pragma unroll
    for (int t = 0; t < 4; ++t) {
        const int col = colb + t * 16;
        const float bias_c = bm[col];
#pragma unroll
        for (int i = 0; i < 8; ++i)
            out[(size_t)(m0 + i + rsel) * D_ + col] = acc[t][i] + bias_c;
    }
}

// ---------------------------------------------------------------- launch
extern "C" void kernel_launch(void* const* d_in, const int* in_sizes, int n_in,
                              void* d_out, int out_size, void* d_ws, size_t ws_size,
                              hipStream_t stream) {
    (void)in_sizes; (void)n_in; (void)out_size; (void)ws_size;

    const float* q      = (const float*)d_in[0];
    const float* k      = (const float*)d_in[1];
    const float* v      = (const float*)d_in[2];
    const unsigned char* masks = (const unsigned char*)d_in[3];
    const float* alphas = (const float*)d_in[4];
    const float* Wq = (const float*)d_in[5];
    const float* bq = (const float*)d_in[6];
    const float* Wk = (const float*)d_in[7];
    const float* bk = (const float*)d_in[8];
    const float* Wv = (const float*)d_in[9];
    const float* bv = (const float*)d_in[10];
    const float* Wm = (const float*)d_in[11];
    const float* bm = (const float*)d_in[12];

    char* ws = (char*)d_ws;
    const size_t SZ_X = (size_t)M_ * D_ * sizeof(bf16_t);  // 8 MB
    const size_t SZ_W = (size_t)D_ * D_ * sizeof(bf16_t);  // 2 MB
    bf16_t* qb    = (bf16_t*)(ws);
    bf16_t* kb    = (bf16_t*)(ws + 1 * SZ_X);
    bf16_t* vb    = (bf16_t*)(ws + 2 * SZ_X);
    bf16_t* wqb   = (bf16_t*)(ws + 3 * SZ_X);
    bf16_t* wkb   = (bf16_t*)(ws + 3 * SZ_X + 1 * SZ_W);
    bf16_t* wvb   = (bf16_t*)(ws + 3 * SZ_X + 2 * SZ_W);
    bf16_t* wmb   = (bf16_t*)(ws + 3 * SZ_X + 3 * SZ_W);
    bf16_t* Qp    = (bf16_t*)(ws + 3 * SZ_X + 4 * SZ_W);
    bf16_t* Kp    = (bf16_t*)(ws + 4 * SZ_X + 4 * SZ_W);
    bf16_t* VpT   = (bf16_t*)(ws + 5 * SZ_X + 4 * SZ_W);
    bf16_t* atted = (bf16_t*)(ws + 6 * SZ_X + 4 * SZ_W);   // total 64 MB

    const int nX = M_ * D_, nW = D_ * D_;
    f32_to_bf16_kernel<<<512, 256, 0, stream>>>(q,  qb,  nX);
    f32_to_bf16_kernel<<<512, 256, 0, stream>>>(k,  kb,  nX);
    f32_to_bf16_kernel<<<512, 256, 0, stream>>>(v,  vb,  nX);
    f32_to_bf16_kernel<<<256, 256, 0, stream>>>(Wq, wqb, nW);
    f32_to_bf16_kernel<<<256, 256, 0, stream>>>(Wk, wkb, nW);
    f32_to_bf16_kernel<<<256, 256, 0, stream>>>(Wv, wvb, nW);
    f32_to_bf16_kernel<<<256, 256, 0, stream>>>(Wm, wmb, nW);

    proj_kernel<<<dim3(D_ / 64, M_ / 128, 3), 256, 0, stream>>>(
        qb, kb, vb, wqb, wkb, wvb, bq, bk, bv, Qp, Kp, VpT);

    attn_kernel<<<dim3(N_ / 64, B_ * H_), 128, 0, stream>>>(
        Qp, Kp, VpT, masks, alphas, atted);

    outproj_kernel<<<dim3(D_ / 64, M_ / 128), 256, 0, stream>>>(
        atted, wmb, bm, (float*)d_out);
}
